// KNRM_32530082300530
// MI455X (gfx1250) — matrix operand order repaired and checked by reading, not compiled, not run on hardware
//
#include <hip/hip_runtime.h>
#include <hip/hip_bf16.h>

typedef __attribute__((ext_vector_type(16))) _Float16 v16h;
typedef __attribute__((ext_vector_type(8)))  _Float16 v8h;
typedef __attribute__((ext_vector_type(8)))  float    v8f;

#define NK       11
#define EMB_DIM  100
#define EMB_PAD  128     // pad 100 -> 128 = 4 x K32 WMMA steps
#define NQ       32
#define ND       2048
#define NB       64
#define DCHUNK   128     // docs per block (8 waves x 16-doc tile)
#define NCHUNKS  (ND / DCHUNK)   // 16

__constant__ float c_mu[NK]  = {-0.9f,-0.7f,-0.5f,-0.3f,-0.1f,0.1f,0.3f,0.5f,0.7f,0.9f,1.0f};
// 0.5 / std^2 :  std = 0.1 -> 50 ;  std = 0.001 (exact kernel) -> 5e5
__constant__ float c_i2s[NK] = {50.f,50.f,50.f,50.f,50.f,50.f,50.f,50.f,50.f,50.f,500000.f};

// ---------------------------------------------------------------------------
// Kernel 1: gather + L2-normalize embeddings, emit f16 rows padded to 128.
// One wave (32 lanes) per token; lane l covers emb indices l, l+32, l+64, l+96.
// ---------------------------------------------------------------------------
__global__ void embed_norm_kernel(const int* __restrict__ qtok,
                                  const int* __restrict__ dtok,
                                  const float* __restrict__ emb,
                                  _Float16* __restrict__ Qn,
                                  _Float16* __restrict__ Dn) {
    const int wave = threadIdx.x >> 5;
    const int lane = threadIdx.x & 31;
    const long t = (long)blockIdx.x * 8 + wave;          // global token id
    const long NTOK = (long)NB * NQ + (long)NB * ND;     // 2048 + 131072
    if (t >= NTOK) return;

    int tok;
    _Float16* outp;
    if (t < (long)NB * NQ) {
        tok  = qtok[t];
        outp = Qn + t * EMB_PAD;
    } else {
        const long td = t - (long)NB * NQ;
        tok  = dtok[td];
        outp = Dn + td * EMB_PAD;
    }

    const float* e = emb + (size_t)tok * EMB_DIM;
    float v[4];
    float ss = 0.0f;
#pragma unroll
    for (int j = 0; j < 4; ++j) {
        const int idx = lane + 32 * j;
        v[j] = (idx < EMB_DIM) ? e[idx] : 0.0f;
        ss += v[j] * v[j];
    }
#pragma unroll
    for (int off = 16; off > 0; off >>= 1)
        ss += __shfl_xor(ss, off, 32);

    const float inv = 1.0f / (sqrtf(ss) + 1e-10f);   // e / (||e|| + 1e-10)
#pragma unroll
    for (int j = 0; j < 4; ++j)
        outp[lane + 32 * j] = (_Float16)(v[j] * inv);
}

// ---------------------------------------------------------------------------
// Kernel 2: per (batch, 128-doc chunk): f16 WMMA GEMM 32 x 128 x 128(K) and
// fused 11-kernel RBF + doc-axis sum into per-block partials.
//
// WMMA fragment construction (ISA 7.12.2, wave32):
//   A (16x32 f16): lane l -> row M = l&15 ; VGPR0..3 = K0+(l>>4)*8 .. +7,
//                  VGPR4..7 = K0+16+(l>>4)*8 .. +7   (two 16B contiguous loads)
//   B (32x16 f16): lane l -> col N = l&15 ; same K order as A
//                  (two 16B contiguous loads)
//   C (16x16 f32): VGPR i : lanes 0-15 -> M=i, N=lane ; lanes 16-31 -> M=i+8.
// ---------------------------------------------------------------------------
__global__ void knrm_gemm_kernel(const _Float16* __restrict__ Qn,
                                 const _Float16* __restrict__ Dn,
                                 float* __restrict__ part) {
    const int b     = blockIdx.x;          // batch
    const int chunk = blockIdx.y;          // 0..15
    const int wave  = threadIdx.x >> 5;    // 0..7, one 16-doc N-tile each
    const int lane  = threadIdx.x & 31;

    __shared__ float acc[NQ * NK];         // qt_match partial for this block
    for (int i = threadIdx.x; i < NQ * NK; i += blockDim.x) acc[i] = 0.0f;
    __syncthreads();

    const _Float16* Qb = Qn + (size_t)b * NQ * EMB_PAD;
    const _Float16* Db = Dn + ((size_t)b * ND + (size_t)chunk * DCHUNK
                               + (size_t)wave * 16) * EMB_PAD;

    const int mA    = lane & 15;
    const int koffA = (lane >> 4) * 8;
    const int nB    = lane & 15;
    const int koffB = (lane >> 4) * 16;

    v8f c0 = {};   // queries 0..15
    v8f c1 = {};   // queries 16..31
#pragma unroll
    for (int ks = 0; ks < 4; ++ks) {
        const int K0 = ks * 32;

        const v8h a0lo = *(const v8h*)(Qb + (size_t)mA * EMB_PAD + K0 + koffA);
        const v8h a0hi = *(const v8h*)(Qb + (size_t)mA * EMB_PAD + K0 + 16 + koffA);
        const v8h a1lo = *(const v8h*)(Qb + (size_t)(16 + mA) * EMB_PAD + K0 + koffA);
        const v8h a1hi = *(const v8h*)(Qb + (size_t)(16 + mA) * EMB_PAD + K0 + 16 + koffA);
        const v16h a0 = __builtin_shufflevector(a0lo, a0hi,
                            0,1,2,3,4,5,6,7,8,9,10,11,12,13,14,15);
        const v16h a1 = __builtin_shufflevector(a1lo, a1hi,
                            0,1,2,3,4,5,6,7,8,9,10,11,12,13,14,15);
        const v16h bf = __builtin_shufflevector(*(const v8h*)(Db + (size_t)nB * EMB_PAD + K0 + koffA), *(const v8h*)(Db + (size_t)nB * EMB_PAD + K0 + 16 + koffA), 0,1,2,3,4,5,6,7,8,9,10,11,12,13,14,15);

        c0 = __builtin_amdgcn_wmma_f32_16x16x32_f16(false, a0, false, bf,
                                                    (short)0, c0, false, false);
        c1 = __builtin_amdgcn_wmma_f32_16x16x32_f16(false, a1, false, bf,
                                                    (short)0, c1, false, false);
    }

    // RBF kernels + sum over docs (N axis) -> LDS (ds_add_f32)
    const int mbase = (lane >> 4) * 8;
#pragma unroll
    for (int i = 0; i < 8; ++i) {
        const float s0 = c0[i];
        const float s1 = c1[i];
        const int m0 = mbase + i;        // query row, tile 0
        const int m1 = 16 + mbase + i;   // query row, tile 1
#pragma unroll
        for (int k = 0; k < NK; ++k) {
            const float d0 = s0 - c_mu[k];
            const float d1 = s1 - c_mu[k];
            atomicAdd(&acc[m0 * NK + k], __expf(-(d0 * d0) * c_i2s[k]));
            atomicAdd(&acc[m1 * NK + k], __expf(-(d1 * d1) * c_i2s[k]));
        }
    }
    __syncthreads();

    // per-block partial, reduced deterministically in kernel 3
    float* p = part + ((size_t)b * NCHUNKS + chunk) * (NQ * NK);
    for (int i = threadIdx.x; i < NQ * NK; i += blockDim.x)
        p[i] = acc[i];
}

// ---------------------------------------------------------------------------
// Kernel 3: per batch: sum 16 chunk partials (fixed order), log1p over docs-
// sum, sum over queries, weight, wave-reduce, + 1e-10.
// ---------------------------------------------------------------------------
__global__ void knrm_reduce_kernel(const float* __restrict__ part,
                                   const float* __restrict__ w,
                                   float* __restrict__ out) {
    const int b    = blockIdx.x;
    const int lane = threadIdx.x;   // 32 threads
    float partial = 0.0f;
    if (lane < NK) {
        const float wl = w[lane];
        float s = 0.0f;
        for (int q = 0; q < NQ; ++q) {
            float qt = 0.0f;
            for (int ch = 0; ch < NCHUNKS; ++ch)
                qt += part[((size_t)b * NCHUNKS + ch) * (NQ * NK) + q * NK + lane];
            s += log1pf(qt);
        }
        partial = s * wl;
    }
#pragma unroll
    for (int off = 16; off > 0; off >>= 1)
        partial += __shfl_xor(partial, off, 32);
    if (lane == 0) out[b] = partial + 1e-10f;
}

// ---------------------------------------------------------------------------
extern "C" void kernel_launch(void* const* d_in, const int* in_sizes, int n_in,
                              void* d_out, int out_size, void* d_ws, size_t ws_size,
                              hipStream_t stream) {
    const int*   qtok = (const int*)d_in[0];     // (64,32,1) int32
    const int*   dtok = (const int*)d_in[1];     // (64,2048,1) int32
    // d_in[2], d_in[3] = qlen, dlen : unused by the reference
    const float* emb  = (const float*)d_in[4];   // (400000,100) f32
    const float* w    = (const float*)d_in[5];   // (1,11) f32
    float*       out  = (float*)d_out;           // (64,) f32

    char* ws = (char*)d_ws;
    const size_t qn_bytes   = (size_t)NB * NQ * EMB_PAD * sizeof(_Float16); //  512 KB
    const size_t dn_bytes   = (size_t)NB * ND * EMB_PAD * sizeof(_Float16); //   32 MB
    _Float16* Qn   = (_Float16*)(ws);
    _Float16* Dn   = (_Float16*)(ws + qn_bytes);
    float*    part = (float*)(ws + qn_bytes + dn_bytes);  // 64*16*352 f32 ≈ 1.4 MB

    // 1) gather + normalize -> f16 padded rows
    {
        const long ntok = (long)NB * NQ + (long)NB * ND;   // 133120
        const int nblk  = (int)((ntok + 7) / 8);           // 8 waves/block
        embed_norm_kernel<<<nblk, 256, 0, stream>>>(qtok, dtok, emb, Qn, Dn);
    }

    // 2) WMMA GEMM + fused RBF kernels + doc-sum partials
    {
        dim3 grid(NB, NCHUNKS);
        knrm_gemm_kernel<<<grid, 256, 0, stream>>>(Qn, Dn, part);
    }

    // 3) deterministic final reduction -> scores
    knrm_reduce_kernel<<<NB, 32, 0, stream>>>(part, w, out);
}
